// GroupedQueryAttention_70205535421087
// MI455X (gfx1250) — matrix operand order repaired and checked by reading, not compiled, not run on hardware
//
#include <hip/hip_runtime.h>

// ---------------------------------------------------------------- constants
#define N_EMBD   2048
#define N_HEADS  32
#define N_KVH    8
#define HEAD_DIM 64
#define N_REP    4          // N_HEADS / N_KVH
#define BSZ      2
#define TSEQ     2048
#define KV_DIM   (N_KVH * HEAD_DIM)   // 512
#define MROWS    (BSZ * TSEQ)         // 4096

#define LDS_PITCH 40        // 32 bf16 row + 8 bf16 TDM pad -> 80B, bank-conflict free

// ---------------------------------------------------------------- types
typedef __attribute__((ext_vector_type(16))) __bf16 bf16x16;
typedef __attribute__((ext_vector_type(8)))  __bf16 bf16x8;
typedef __attribute__((ext_vector_type(8)))  float  f32x8;
typedef __attribute__((ext_vector_type(4)))  unsigned int u32x4;
typedef __attribute__((ext_vector_type(8)))  int i32x8;
typedef __attribute__((ext_vector_type(4)))  int i32x4;

union BF16x16 { bf16x16 v; bf16x8 h[2]; __bf16 e[16]; };
union F32x8   { f32x8 v;  float e[8]; };

static __device__ __forceinline__ f32x8 wmma_bf16(bf16x16 a, bf16x16 b, f32x8 c) {
  // D = A(16x32 bf16) * B(32x16 bf16) + C(16x16 f32)
  return __builtin_amdgcn_wmma_f32_16x16x32_bf16(false, a, false, b,
                                                 (short)0, c, false, false);
}

// A-fragment (16x32 bf16, row-major source, lane holds row = lane&15):
// element e -> K = half*16 + e (contiguous)
static __device__ __forceinline__ bf16x16 load_a_frag(const __bf16* row_k0, int half) {
  BF16x16 r;
  r.h[0] = *(const bf16x8*)(row_k0 + half * 16);
  r.h[1] = *(const bf16x8*)(row_k0 + half * 16 + 8);
  return r.v;
}

// B-fragment when the source is K-major per output column (pre-transposed):
// element e -> K = e + half*16, N fixed per lane -> 32B contiguous per lane.
static __device__ __forceinline__ bf16x16 load_b_frag_kmajor(const __bf16* col_k0, int half) {
  return *(const bf16x16*)(col_k0 + half * 16);
}
// Same fragment but from 16B-aligned LDS (80B pitch rows): two b64x2 chunks.
static __device__ __forceinline__ bf16x16 load_b_frag_lds(const __bf16* col_k0, int half) {
  BF16x16 r;
  r.h[0] = *(const bf16x8*)(col_k0 + half * 16);
  r.h[1] = *(const bf16x8*)(col_k0 + half * 16 + 8);
  return r.v;
}

// ---------------------------------------------------------------- TDM
// Issue a 2-D tensor_load_to_lds: tile_rows x 32 bf16, row stride = ld (elems),
// padded by TDM to LDS_PITCH (pad 4 DWORDs after every 16 DWORDs stored).
static __device__ __forceinline__ void tdm_load_panel(unsigned int lds_off,
                                                      const __bf16* gptr,
                                                      int tile_rows, int ld) {
  unsigned long long ga = (unsigned long long)(uintptr_t)gptr;
  u32x4 g0;
  g0[0] = 1u;                                        // count=1 (valid user D#)
  g0[1] = lds_off;                                   // lds_addr (bytes)
  g0[2] = (unsigned int)(ga & 0xffffffffu);          // global_addr[31:0]
  g0[3] = (unsigned int)((ga >> 32) & 0x01ffffffu)   // global_addr[56:32]
        | (2u << 30);                                // type = 2 ("image")
  i32x8 g1;
  g1[0] = (int)((1u << 16)       // data_size = 2 bytes
        | (1u << 20)             // pad_enable
        | (3u << 22)             // pad_interval: 16 DWORDs (= one 32-elem row)
        | (3u << 25));           // pad_amount: 4 DWORDs  (row pitch -> 40 elems)
  unsigned int td0 = (unsigned int)ld;               // tensor_dim0
  unsigned int td1 = 0x7fffffffu;                    // tensor_dim1: no clipping
  g1[1] = (int)((td0 & 0xffffu) << 16);              // bits 63:48 td0[15:0]
  g1[2] = (int)((td0 >> 16) | ((td1 & 0xffffu) << 16)); // td0[31:16] | td1[15:0]
  g1[3] = (int)((td1 >> 16) | (32u << 16));          // td1[31:16] | tile_dim0=32
  g1[4] = (int)((unsigned int)tile_rows & 0xffffu);  // tile_dim1; tile_dim2=0
  g1[5] = (int)td0;                                  // tensor_dim0_stride[31:0]
  g1[6] = 0;                                         // stride[47:32]=0
  g1[7] = 0;
  i32x4 z4 = {0, 0, 0, 0};
  i32x8 z8 = {0, 0, 0, 0, 0, 0, 0, 0};
  // 6-arg form (clang-23 / therock-10.0): (g0, g1, g2, g3, extra, cpol)
  __builtin_amdgcn_tensor_load_to_lds(g0, g1, z4, z4, z8, 0);
}

// ---------------------------------------------------------------- conversions
__global__ void cvt_f32_bf16(const float* __restrict__ in, __bf16* __restrict__ out, int n) {
  int i = blockIdx.x * 256 + threadIdx.x;
  if (i < n) out[i] = (__bf16)in[i];
}

// in: [K,N] f32 row-major  ->  out: [N,K] bf16 (W transposed, K contiguous)
__global__ void cvt_f32_bf16_T(const float* __restrict__ in, __bf16* __restrict__ out,
                               int K, int N) {
  int i = blockIdx.x * 256 + threadIdx.x;
  if (i < K * N) {
    int n = i / K, k = i % K;
    out[i] = (__bf16)in[(size_t)k * N + n];
  }
}

// ---------------------------------------------------------------- GEMM
// C[M,N] = A[M,K](bf16,row-major) * Bt[N,K](bf16, pre-transposed) + bias.
// 128x128 block tile, 4 waves (each 64x64 = 16 WMMAs / K-step).
// Operand panels (128x32) staged in LDS by the Tensor Data Mover, double
// buffered, producer wave0 synchronizes via TENSORcnt + workgroup barrier.
// MODE 0: f32 row-major  MODE 1: bf16 row-major  MODE 2: bf16 [B][N][T] (V)
template<int MODE>
__global__ __launch_bounds__(128)
void gemm_bf16_wmma(const __bf16* __restrict__ A, const __bf16* __restrict__ Bt,
                    const float* __restrict__ bias, void* __restrict__ Cout,
                    int M, int N, int K) {
  __shared__ __align__(16) __bf16 ldsA[2][128 * LDS_PITCH];
  __shared__ __align__(16) __bf16 ldsB[2][128 * LDS_PITCH];

  const int tid  = threadIdx.x;
  const int lane = tid & 31;
  const int wave = tid >> 5;
  const int row  = lane & 15;
  const int half = lane >> 4;
  const int mw   = (wave >> 1) * 64;       // wave tile inside block: 2x2 grid
  const int nw   = (wave & 1) * 64;
  const int m0 = blockIdx.y * 128;
  const int n0 = blockIdx.x * 128;

  F32x8 acc[4][4] = {};

  // preload K-step 0
  if (wave == 0) {
    tdm_load_panel((unsigned int)(uintptr_t)(void*)&ldsA[0][0],
                   A + (size_t)m0 * K, 128, K);
    tdm_load_panel((unsigned int)(uintptr_t)(void*)&ldsB[0][0],
                   Bt + (size_t)n0 * K, 128, K);
    __builtin_amdgcn_s_wait_tensorcnt((short)0);
  }
  __syncthreads();

  int cur = 0;
  for (int k0 = 0; k0 < K; k0 += 32) {
    const int nxt = cur ^ 1;
    const bool more = (k0 + 32) < K;
    if (wave == 0 && more) {      // TDM prefetch of next panels (async)
      tdm_load_panel((unsigned int)(uintptr_t)(void*)&ldsA[nxt][0],
                     A + (size_t)m0 * K + (k0 + 32), 128, K);
      tdm_load_panel((unsigned int)(uintptr_t)(void*)&ldsB[nxt][0],
                     Bt + (size_t)n0 * K + (k0 + 32), 128, K);
    }

    bf16x16 a[4], b[4];
#pragma unroll
    for (int i = 0; i < 4; ++i)
      a[i] = load_a_frag(&ldsA[cur][(mw + i * 16 + row) * LDS_PITCH], half);
#pragma unroll
    for (int i = 0; i < 4; ++i)
      b[i] = load_b_frag_lds(&ldsB[cur][(nw + i * 16 + row) * LDS_PITCH], half);
#pragma unroll
    for (int mi = 0; mi < 4; ++mi)
#pragma unroll
      for (int ni = 0; ni < 4; ++ni)
        acc[mi][ni].v = wmma_bf16(a[mi], b[ni], acc[mi][ni].v);

    if (wave == 0 && more) __builtin_amdgcn_s_wait_tensorcnt((short)0);
    __syncthreads();
    cur = nxt;
  }

#pragma unroll
  for (int mi = 0; mi < 4; ++mi) {
#pragma unroll
    for (int ni = 0; ni < 4; ++ni) {
      const int ncol = n0 + nw + ni * 16 + row;
      const float bz = bias[ncol];
#pragma unroll
      for (int v = 0; v < 8; ++v) {
        const int mrow = m0 + mw + mi * 16 + v + 8 * half;
        const float val = acc[mi][ni].e[v] + bz;
        if (MODE == 0) {
          ((float*)Cout)[(size_t)mrow * N + ncol] = val;
        } else if (MODE == 1) {
          ((__bf16*)Cout)[(size_t)mrow * N + ncol] = (__bf16)val;
        } else {
          const int bb = mrow / TSEQ, t = mrow % TSEQ;
          ((__bf16*)Cout)[((size_t)bb * N + ncol) * TSEQ + t] = (__bf16)val;
        }
      }
    }
  }
}

// ---------------------------------------------------------------- attention
// One wave per (b, head, 16-query tile). Flash-style streaming softmax over
// 32-key chunks. QK^T and PV both via v_wmma_f32_16x16x32_bf16.
// q,k: bf16 row-major [B*T, dim]; vT: bf16 [B][KV_DIM][T]; y: bf16 [B*T, N_EMBD]
__global__ __launch_bounds__(32)
void attn_wmma(const __bf16* __restrict__ qb, const __bf16* __restrict__ kb,
               const __bf16* __restrict__ vT, __bf16* __restrict__ yb) {
  __shared__ __align__(32) __bf16 p_lds[16 * 32];

  const int lane = threadIdx.x;
  const int row  = lane & 15;
  const int half = lane >> 4;

  const int qt = blockIdx.x % (TSEQ / 16);
  const int h  = (blockIdx.x / (TSEQ / 16)) % N_HEADS;
  const int b  = blockIdx.x / ((TSEQ / 16) * N_HEADS);
  const int hk = h / N_REP;
  const int q0 = qt * 16;

  const __bf16* qbase = qb + (size_t)(b * TSEQ + q0 + row) * N_EMBD + h * HEAD_DIM;
  bf16x16 qa0 = load_a_frag(qbase, half);
  bf16x16 qa1 = load_a_frag(qbase + 32, half);

  F32x8 o[4] = {};
  float m_run[8], l_run[8];
#pragma unroll
  for (int v = 0; v < 8; ++v) { m_run[v] = -1e30f; l_run[v] = 0.0f; }

  const int kend = q0 + 16;
  for (int kc = 0; kc < kend; kc += 32) {
    if (kc + 32 < kend) {     // stream prefetch of next K/V chunk
      const __bf16* kpf = kb + (size_t)(b * TSEQ + kc + 32 + row) * KV_DIM + hk * HEAD_DIM;
      __builtin_prefetch((const void*)kpf, 0, 3);
      const __bf16* vpf = vT + ((size_t)b * KV_DIM + hk * HEAD_DIM + row) * TSEQ + kc + 32;
      __builtin_prefetch((const void*)vpf, 0, 3);
    }

    // ---- scores: two 16x16 tiles covering keys kc..kc+31
    F32x8 s[2];
#pragma unroll
    for (int t = 0; t < 2; ++t) {
      const __bf16* kp = kb + (size_t)(b * TSEQ + kc + t * 16 + row) * KV_DIM
                            + hk * HEAD_DIM;
      f32x8 z = {};
      z = wmma_bf16(qa0, load_b_frag_kmajor(kp,      half), z);
      z = wmma_bf16(qa1, load_b_frag_kmajor(kp + 32, half), z);
      s[t].v = z;
#pragma unroll
      for (int v = 0; v < 8; ++v) {
        const int qi = q0 + v + 8 * half;
        const int ki = kc + t * 16 + row;
        float val = s[t].e[v] * 0.125f;            // 1/sqrt(64)
        if (ki > qi) val = -1e30f;
        s[t].e[v] = val;
      }
    }

    // ---- online softmax stats (reduce across the 16-lane group)
#pragma unroll
    for (int v = 0; v < 8; ++v) {
      float mx = fmaxf(s[0].e[v], s[1].e[v]);
#pragma unroll
      for (int off = 1; off < 16; off <<= 1) mx = fmaxf(mx, __shfl_xor(mx, off));
      const float m_new = fmaxf(m_run[v], mx);
      const float alpha = __expf(m_run[v] - m_new);
      s[0].e[v] = __expf(s[0].e[v] - m_new);
      s[1].e[v] = __expf(s[1].e[v] - m_new);
      float rs = s[0].e[v] + s[1].e[v];
#pragma unroll
      for (int off = 1; off < 16; off <<= 1) rs += __shfl_xor(rs, off);
      l_run[v] = l_run[v] * alpha + rs;
      m_run[v] = m_new;
#pragma unroll
      for (int dt = 0; dt < 4; ++dt) o[dt].e[v] *= alpha;
    }

    // ---- P: C-layout -> A-layout via LDS (bf16)
#pragma unroll
    for (int v = 0; v < 8; ++v) {
      const int pr = v + 8 * half;
      p_lds[pr * 32 + row]      = (__bf16)s[0].e[v];
      p_lds[pr * 32 + 16 + row] = (__bf16)s[1].e[v];
    }
    asm volatile("s_wait_dscnt 0" ::: "memory");
    BF16x16 pa;
    pa.h[0] = *(const bf16x8*)(&p_lds[row * 32 + half * 16]);
    pa.h[1] = *(const bf16x8*)(&p_lds[row * 32 + half * 16 + 8]);

    // ---- PV: V stored K-major ([B][KV_DIM][T]) -> contiguous b-fragments
#pragma unroll
    for (int dt = 0; dt < 4; ++dt) {
      const __bf16* vp = vT + ((size_t)b * KV_DIM + hk * HEAD_DIM + dt * 16 + row) * TSEQ
                            + kc;
      o[dt].v = wmma_bf16(pa.v, load_b_frag_kmajor(vp, half), o[dt].v);
    }
  }

  // ---- epilogue: y[b,t, h*64 + d] = O / l
#pragma unroll
  for (int dt = 0; dt < 4; ++dt) {
#pragma unroll
    for (int v = 0; v < 8; ++v) {
      const int qi = q0 + v + 8 * half;
      yb[(size_t)(b * TSEQ + qi) * N_EMBD + h * HEAD_DIM + dt * 16 + row] =
          (__bf16)(o[dt].e[v] / l_run[v]);
    }
  }
}

// ---------------------------------------------------------------- launch
extern "C" void kernel_launch(void* const* d_in, const int* in_sizes, int n_in,
                              void* d_out, int out_size, void* d_ws, size_t ws_size,
                              hipStream_t stream) {
  (void)in_sizes; (void)n_in; (void)out_size; (void)ws_size;

  const float* x  = (const float*)d_in[0];
  const float* Wq = (const float*)d_in[1];
  const float* bq = (const float*)d_in[2];
  const float* Wk = (const float*)d_in[3];
  const float* bk = (const float*)d_in[4];
  const float* Wv = (const float*)d_in[5];
  const float* bv = (const float*)d_in[6];
  const float* Wo = (const float*)d_in[7];
  const float* bo = (const float*)d_in[8];
  float* out = (float*)d_out;

  // workspace carve-up (~76 MB total, bf16 buffers)
  char* ws = (char*)d_ws;
  __bf16* xbf  = (__bf16*)ws; ws += (size_t)MROWS * N_EMBD * 2;
  __bf16* wqT  = (__bf16*)ws; ws += (size_t)N_EMBD * N_EMBD * 2;
  __bf16* wkT  = (__bf16*)ws; ws += (size_t)N_EMBD * KV_DIM * 2;
  __bf16* wvT  = (__bf16*)ws; ws += (size_t)N_EMBD * KV_DIM * 2;
  __bf16* woT  = (__bf16*)ws; ws += (size_t)N_EMBD * N_EMBD * 2;
  __bf16* qbf  = (__bf16*)ws; ws += (size_t)MROWS * N_EMBD * 2;
  __bf16* kbf  = (__bf16*)ws; ws += (size_t)MROWS * KV_DIM * 2;
  __bf16* vbfT = (__bf16*)ws; ws += (size_t)MROWS * KV_DIM * 2;
  __bf16* ybf  = (__bf16*)ws;

  // 1) precision conversion (weights transposed so GEMM B-frags are contiguous)
  {
    int n = MROWS * N_EMBD;
    cvt_f32_bf16<<<(n + 255) / 256, 256, 0, stream>>>(x, xbf, n);
  }
  cvt_f32_bf16_T<<<(N_EMBD * N_EMBD + 255) / 256, 256, 0, stream>>>(Wq, wqT, N_EMBD, N_EMBD);
  cvt_f32_bf16_T<<<(N_EMBD * KV_DIM + 255) / 256, 256, 0, stream>>>(Wk, wkT, N_EMBD, KV_DIM);
  cvt_f32_bf16_T<<<(N_EMBD * KV_DIM + 255) / 256, 256, 0, stream>>>(Wv, wvT, N_EMBD, KV_DIM);
  cvt_f32_bf16_T<<<(N_EMBD * N_EMBD + 255) / 256, 256, 0, stream>>>(Wo, woT, N_EMBD, N_EMBD);

  // 2) projections (TDM-staged WMMA bf16, f32 accumulate, bias fused)
  gemm_bf16_wmma<1><<<dim3(N_EMBD / 128, MROWS / 128), 128, 0, stream>>>(
      xbf, wqT, bq, qbf, MROWS, N_EMBD, N_EMBD);
  gemm_bf16_wmma<1><<<dim3(KV_DIM / 128, MROWS / 128), 128, 0, stream>>>(
      xbf, wkT, bk, kbf, MROWS, KV_DIM, N_EMBD);
  gemm_bf16_wmma<2><<<dim3(KV_DIM / 128, MROWS / 128), 128, 0, stream>>>(
      xbf, wvT, bv, vbfT, MROWS, KV_DIM, N_EMBD);   // V stored [B][KV_DIM][T]

  // 3) flash attention (causal, GQA 4:1)
  attn_wmma<<<BSZ * N_HEADS * (TSEQ / 16), 32, 0, stream>>>(qbf, kbf, vbfT, ybf);

  // 4) output projection -> f32 out
  gemm_bf16_wmma<0><<<dim3(N_EMBD / 128, MROWS / 128), 128, 0, stream>>>(
      ybf, woT, bo, out, MROWS, N_EMBD, N_EMBD);
}
